// RSSM_75565654605852
// MI455X (gfx1250) — matrix operand order repaired and checked by reading, not compiled, not run on hardware
//
#include <hip/hip_runtime.h>
#include <hip/hip_bf16.h>
#include <stdint.h>

// ---------------------------------------------------------------------------
// RSSM forward for MI455X (gfx1250, wave32).
// - All big GEMMs on v_wmma_f32_16x16x32_bf16 (bf16 weights converted once,
//   transposed [Npad][Kpad], zero padded; ~56 MB -> L2 resident).
// - GEMM blocks: 8 waves cover 8 M-tiles (128 rows) x one 64-col N slab.
//   B is staged in double-buffered LDS, 64 K per buffer (two 32-K chunks,
//   row stride padded to 40 bf16 -> bank-conflict-free fragment reads).
//   Fill is unconditional with a peeled final iteration (branchless hot loop,
//   one workgroup barrier per 64 K).
// - Scan GEMMs are K-split into 4 partial launches (disjoint f32 partial
//   buffers); consumers (GRU gates, LayerNorm, posts-sum) fold the partial
//   reduction + bias for free.
// ---------------------------------------------------------------------------

typedef __bf16 bf16_t;
typedef __attribute__((ext_vector_type(16))) __bf16 v16bf;
typedef __attribute__((ext_vector_type(8)))  __bf16 v8bf;
typedef __attribute__((ext_vector_type(8)))  float  v8f;

#define T_DIM 64
#define B_DIM 128
#define E_DIM 1536
#define A_DIM 6
#define D_DIM 2048
#define H_DIM 1000
#define HP_DIM 1024
#define SK_DIM 1024
#define TB_DIM (T_DIM * B_DIM)
#define FEAT_DIM (D_DIM + SK_DIM)

#define LDSB_STRIDE 40              // 32 k-elems + 8 pad
#define LDSB_CHUNK  (64 * LDSB_STRIDE)   // one 64N x 32K chunk

__device__ __forceinline__ bf16_t f2bf(float f) {
    uint32_t u = __builtin_bit_cast(uint32_t, f);
    uint32_t r = (u + 0x7FFFu + ((u >> 16) & 1u)) >> 16;   // RNE
    uint16_t s = (uint16_t)r;
    return __builtin_bit_cast(bf16_t, s);
}

// ---------------------------------------------------------------------------
// bf16 WMMA GEMM, LDS-staged B (double buffered, 64 K per buffer).
//   A [M,K] bf16 row-major (lda), Bt = W^T [Npad,K] bf16 row-major (ldb),
//   C [M,ldc] f32.  M multiple of 128, K multiple of 64.
// Block: 256 threads = 8 waves -> (128-row M group) x (64-col N slab); wave w
// owns M-tile w; B slab shared via LDS. Fragment layouts per CDNA5 ISA 7.12.2.
// ---------------------------------------------------------------------------
__global__ __launch_bounds__(256)
void gemm_bf16_wmma(const bf16_t* __restrict__ A, int lda,
                    const bf16_t* __restrict__ Bt, int ldb,
                    const float* __restrict__ bias,
                    float* __restrict__ C, int ldc,
                    int Nvalid, int Ntiles64, int K) {
    __shared__ bf16_t smB[4 * LDSB_CHUNK];   // 2 buffers x 2 chunks

    const int mg = blockIdx.x / Ntiles64;
    const int tn = blockIdx.x % Ntiles64;
    const int w    = threadIdx.x >> 5;
    const int lane = threadIdx.x & 31;
    const int lo = lane & 15;
    const int hi = lane >> 4;
    const int m0 = mg * 128 + w * 16;
    const int n0 = tn << 6;

    const int fill_n = threadIdx.x >> 2;          // 0..63
    const int fill_k = (threadIdx.x & 3) << 3;    // 0,8,16,24
    const bf16_t* __restrict__ fill_src = Bt + (size_t)(n0 + fill_n) * ldb + fill_k;
    bf16_t* const fill_dst = &smB[fill_n * LDSB_STRIDE + fill_k];

    v8f acc[4] = {v8f{}, v8f{}, v8f{}, v8f{}};
    const bf16_t* __restrict__ arow = A + (size_t)(m0 + lo) * lda;

    // prologue: buffer 0 <- K[0,64)
    *(v8bf*)(fill_dst)              = *(const v8bf*)(fill_src);
    *(v8bf*)(fill_dst + LDSB_CHUNK) = *(const v8bf*)(fill_src + 32);

    int buf = 0;
    int k0 = 0;
    for (; k0 + 64 < K; k0 += 64) {
        __syncthreads();
        // unconditional prefetch of next 64 K into the other buffer
        {
            bf16_t* d = fill_dst + (buf ^ 1) * (2 * LDSB_CHUNK);
            *(v8bf*)(d)              = *(const v8bf*)(fill_src + k0 + 64);
            *(v8bf*)(d + LDSB_CHUNK) = *(const v8bf*)(fill_src + k0 + 96);
        }
        #pragma unroll
        for (int c = 0; c < 2; ++c) {
            const int kk = k0 + 32 * c;
            v8bf alo = *(const v8bf*)(arow + kk + 8 * hi);
            v8bf ahi = *(const v8bf*)(arow + kk + 16 + 8 * hi);
            v16bf af = __builtin_shufflevector(alo, ahi,
                        0,1,2,3,4,5,6,7,8,9,10,11,12,13,14,15);
            const bf16_t* bufp = &smB[buf * (2 * LDSB_CHUNK) + c * LDSB_CHUNK];
            v16bf bfv[4];
            #pragma unroll
            for (int sub = 0; sub < 4; ++sub) {
                const bf16_t* bl = bufp + (sub * 16 + lo) * LDSB_STRIDE + 8 * hi;
                v8bf b0 = *(const v8bf*)(bl);
                v8bf b1 = *(const v8bf*)(bl + 16);
                bfv[sub] = __builtin_shufflevector(b0, b1,
                            0,1,2,3,4,5,6,7,8,9,10,11,12,13,14,15);
            }
            #pragma unroll
            for (int sub = 0; sub < 4; ++sub)
                acc[sub] = __builtin_amdgcn_wmma_f32_16x16x32_bf16(false, af, false,
                            bfv[sub], (short)0, acc[sub], false, false);
        }
        buf ^= 1;
    }
    // peeled final 64 K (no fill)
    __syncthreads();
    #pragma unroll
    for (int c = 0; c < 2; ++c) {
        const int kk = k0 + 32 * c;
        v8bf alo = *(const v8bf*)(arow + kk + 8 * hi);
        v8bf ahi = *(const v8bf*)(arow + kk + 16 + 8 * hi);
        v16bf af = __builtin_shufflevector(alo, ahi,
                    0,1,2,3,4,5,6,7,8,9,10,11,12,13,14,15);
        const bf16_t* bufp = &smB[buf * (2 * LDSB_CHUNK) + c * LDSB_CHUNK];
        v16bf bfv[4];
        #pragma unroll
        for (int sub = 0; sub < 4; ++sub) {
            const bf16_t* bl = bufp + (sub * 16 + lo) * LDSB_STRIDE + 8 * hi;
            v8bf b0 = *(const v8bf*)(bl);
            v8bf b1 = *(const v8bf*)(bl + 16);
            bfv[sub] = __builtin_shufflevector(b0, b1,
                        0,1,2,3,4,5,6,7,8,9,10,11,12,13,14,15);
        }
        #pragma unroll
        for (int sub = 0; sub < 4; ++sub)
            acc[sub] = __builtin_amdgcn_wmma_f32_16x16x32_bf16(false, af, false,
                        bfv[sub], (short)0, acc[sub], false, false);
    }

    const int mrow = m0 + 8 * hi;
    #pragma unroll
    for (int sub = 0; sub < 4; ++sub) {
        int n = n0 + sub * 16 + lo;
        if (n >= Nvalid) continue;
        float bv = bias ? bias[n] : 0.0f;
        #pragma unroll
        for (int r = 0; r < 8; ++r)
            C[(size_t)(mrow + r) * ldc + n] = acc[sub][r] + bv;
    }
}

// ---------------------------------------------------------------------------
// Conversions
// ---------------------------------------------------------------------------
__global__ void convert_weight_t(const float* __restrict__ W, bf16_t* __restrict__ Wt,
                                 int K, int N, int Kpad, int Npad) {
    int idx = blockIdx.x * blockDim.x + threadIdx.x;
    if (idx >= Npad * Kpad) return;
    int n = idx / Kpad, k = idx % Kpad;
    float v = (k < K && n < N) ? W[(size_t)k * N + n] : 0.0f;
    Wt[idx] = f2bf(v);
}

__global__ void convert_f32_bf16(const float* __restrict__ src, bf16_t* __restrict__ dst,
                                 int n) {
    int idx = blockIdx.x * blockDim.x + threadIdx.x;
    if (idx < n) dst[idx] = f2bf(src[idx]);
}

__global__ void copy_f32(const float* __restrict__ src, float* __restrict__ dst, int n) {
    int idx = blockIdx.x * blockDim.x + threadIdx.x;
    if (idx < n) dst[idx] = src[idx];
}

__global__ void gather_h_bf16(const float* __restrict__ feat, bf16_t* __restrict__ dst) {
    int idx = blockIdx.x * blockDim.x + threadIdx.x;
    if (idx >= TB_DIM * D_DIM) return;
    int row = idx / D_DIM, j = idx % D_DIM;
    dst[idx] = f2bf(feat[(size_t)row * FEAT_DIM + j]);
}

// ---------------------------------------------------------------------------
// Per-step prep: reset mask on h (f32, in place), bf16 h & z.
// ---------------------------------------------------------------------------
__global__ void prep_step(const unsigned char* __restrict__ reset_t,
                          float* __restrict__ h, bf16_t* __restrict__ h_bf,
                          const float* __restrict__ z_src, bf16_t* __restrict__ z_bf) {
    int idx = blockIdx.x * blockDim.x + threadIdx.x;
    if (idx >= B_DIM * D_DIM) return;
    int b = idx / D_DIM;
    float m = reset_t[b] ? 0.0f : 1.0f;
    float hv = h[idx] * m;
    h[idx] = hv;
    h_bf[idx] = f2bf(hv);
    if (idx < B_DIM * SK_DIM) {
        int b2 = idx / SK_DIM;
        float m2 = reset_t[b2] ? 0.0f : 1.0f;
        z_bf[idx] = f2bf(z_src[idx] * m2);
    }
}

// ---------------------------------------------------------------------------
// Sum-of-partials + bias + LayerNorm(eps=1e-3) + ELU -> bf16 (zero padded),
// with optional fused a@a_w (K=6) term. One 256-thread block per row.
// ---------------------------------------------------------------------------
__global__ __launch_bounds__(256)
void ln_elu_bf16(const float* __restrict__ parts, int P, size_t pstride, int ld,
                 const float* __restrict__ addb, int Hv,
                 const float* __restrict__ scale, const float* __restrict__ bias,
                 bf16_t* __restrict__ Out, int ldo,
                 const float* __restrict__ act, const float* __restrict__ aW) {
    __shared__ float row[HP_DIM];
    __shared__ float red[2][8];
    int b = blockIdx.x;
    float a6[A_DIM];
    if (act) {
        #pragma unroll
        for (int j = 0; j < A_DIM; ++j) a6[j] = act[(size_t)b * A_DIM + j];
    }
    float s = 0.f, s2 = 0.f;
    for (int i = threadIdx.x; i < Hv; i += blockDim.x) {
        float v = addb ? addb[i] : 0.0f;
        for (int p = 0; p < P; ++p)
            v += parts[(size_t)p * pstride + (size_t)b * ld + i];
        if (act) {
            #pragma unroll
            for (int j = 0; j < A_DIM; ++j) v += a6[j] * aW[(size_t)j * Hv + i];
        }
        row[i] = v; s += v; s2 += v * v;
    }
    #pragma unroll
    for (int o = 16; o > 0; o >>= 1) {
        s  += __shfl_down(s,  o, 32);
        s2 += __shfl_down(s2, o, 32);
    }
    int w = threadIdx.x >> 5;
    if ((threadIdx.x & 31) == 0) { red[0][w] = s; red[1][w] = s2; }
    __syncthreads();
    if (threadIdx.x == 0) {
        float ts = 0.f, t2 = 0.f;
        #pragma unroll
        for (int i = 0; i < 8; ++i) { ts += red[0][i]; t2 += red[1][i]; }
        red[0][0] = ts / (float)Hv;
        red[1][0] = t2 / (float)Hv;
    }
    __syncthreads();
    float mean = red[0][0];
    float var  = red[1][0] - mean * mean;
    float inv  = rsqrtf(var + 1e-3f);
    for (int i = threadIdx.x; i < ldo; i += blockDim.x) {
        float o = 0.0f;
        if (i < Hv) {
            float v = (row[i] - mean) * inv * scale[i] + bias[i];
            o = (v > 0.0f) ? v : expm1f(v);   // ELU(alpha=1)
        }
        Out[(size_t)b * ldo + i] = f2bf(o);
    }
}

// ---------------------------------------------------------------------------
// GRUCell gate math; sums 4 gi partials (parts[0..3]) and 4 gh partials
// (parts[4..7]) + biases; writes new h (f32 + bf16) and features[:, :D].
// ---------------------------------------------------------------------------
__global__ void gru_gates(const float* __restrict__ parts, size_t pstride,
                          const float* __restrict__ bi, const float* __restrict__ bh,
                          float* __restrict__ h, bf16_t* __restrict__ h_bf,
                          float* __restrict__ feat_t) {
    int idx = blockIdx.x * blockDim.x + threadIdx.x;
    if (idx >= B_DIM * D_DIM) return;
    int b = idx / D_DIM, j = idx % D_DIM;
    size_t base = (size_t)b * 3 * D_DIM;
    float ir = bi[j], iz = bi[D_DIM + j], in = bi[2 * D_DIM + j];
    float hr = bh[j], hz = bh[D_DIM + j], hn = bh[2 * D_DIM + j];
    #pragma unroll
    for (int p = 0; p < 4; ++p) {
        const float* gp = parts + (size_t)p * pstride + base;
        ir += gp[j]; iz += gp[D_DIM + j]; in += gp[2 * D_DIM + j];
        const float* hp = parts + (size_t)(4 + p) * pstride + base;
        hr += hp[j]; hz += hp[D_DIM + j]; hn += hp[2 * D_DIM + j];
    }
    float r = 1.0f / (1.0f + expf(-(ir + hr)));
    float u = 1.0f / (1.0f + expf(-(iz + hz)));
    float n = tanhf(in + r * hn);
    float hnew = (1.0f - u) * n + u * h[idx];
    h[idx] = hnew;
    h_bf[idx] = f2bf(hnew);
    feat_t[(size_t)b * FEAT_DIM + j] = hnew;
}

// posts[t] = sum of 4 partials + bias
__global__ void sum_bias(const float* __restrict__ parts, size_t pstride,
                         const float* __restrict__ bias, float* __restrict__ out) {
    int idx = blockIdx.x * blockDim.x + threadIdx.x;
    if (idx >= B_DIM * SK_DIM) return;
    int n = idx & (SK_DIM - 1);
    float v = bias[n];
    #pragma unroll
    for (int p = 0; p < 4; ++p) v += parts[(size_t)p * pstride + idx];
    out[idx] = v;
}

// ---------------------------------------------------------------------------
// Gumbel-max categorical -> one-hot. One wave per (b,s) group of 32.
// ---------------------------------------------------------------------------
__device__ __forceinline__ float hash_uniform(uint32_t x) {
    x *= 2654435761u; x ^= x >> 17; x *= 0xed5ad4bbu; x ^= x >> 11;
    x *= 0xac4c1b51u; x ^= x >> 15; x *= 0x31848babu; x ^= x >> 14;
    return ((float)(x >> 8) + 0.5f) * (1.0f / 16777216.0f);
}

__global__ __launch_bounds__(256)
void gumbel_sample(const float* __restrict__ post_t, int t,
                   float* __restrict__ samp_t, float* __restrict__ feat_t) {
    int g = blockIdx.x * (blockDim.x >> 5) + (threadIdx.x >> 5);
    if (g >= B_DIM * 32) return;
    int lane = threadIdx.x & 31;
    int b = g >> 5, s = g & 31;
    int idx = b * SK_DIM + s * 32 + lane;
    uint32_t id = (uint32_t)(((t * B_DIM + b) * 32 + s) * 32 + lane) ^ 0x9E3779B9u;
    float u = hash_uniform(id);
    float v = post_t[idx] + (-logf(-logf(u)));
    int best = lane;
    float bv = v;
    #pragma unroll
    for (int o = 16; o > 0; o >>= 1) {
        float ov = __shfl_xor(bv, o, 32);
        int   ob = __shfl_xor(best, o, 32);
        if (ov > bv || (ov == bv && ob < best)) { bv = ov; best = ob; }
    }
    float one = (lane == best) ? 1.0f : 0.0f;
    samp_t[idx] = one;
    feat_t[(size_t)b * FEAT_DIM + D_DIM + s * 32 + lane] = one;
}

// ---------------------------------------------------------------------------
// Host orchestration
// ---------------------------------------------------------------------------
static inline size_t align256(size_t x) { return (x + 255) & ~(size_t)255; }

static inline void launch_gemm(const bf16_t* A, int lda, const bf16_t* Bt, int ldb,
                               const float* bias, float* C, int ldc,
                               int M, int Nvalid, int Npad, int K, hipStream_t s) {
    int ntiles = Npad >> 6;
    int blocks = (M >> 7) * ntiles;
    gemm_bf16_wmma<<<blocks, 256, 0, s>>>(A, lda, Bt, ldb, bias, C, ldc,
                                          Nvalid, ntiles, K);
}

static inline void launch_gemm_split(const bf16_t* A, int lda, const bf16_t* Bt, int ldb,
                                     float* part, size_t pstride, int pbase, int P,
                                     int N, int Npad, int K, int M, hipStream_t s) {
    int Kc = K / P;    // multiple of 64 for all call sites
    int ntiles = Npad >> 6;
    int blocks = (M >> 7) * ntiles;
    for (int p = 0; p < P; ++p)
        gemm_bf16_wmma<<<blocks, 256, 0, s>>>(A + p * Kc, lda, Bt + p * Kc, ldb, nullptr,
                                              part + (size_t)(pbase + p) * pstride, N,
                                              Npad, ntiles, Kc);
}

extern "C" void kernel_launch(void* const* d_in, const int* in_sizes, int n_in,
                              void* d_out, int out_size, void* d_ws, size_t ws_size,
                              hipStream_t stream) {
    const float* embeds = (const float*)d_in[0];
    const float* action = (const float*)d_in[1];
    const unsigned char* reset = (const unsigned char*)d_in[2];
    const float* h0 = (const float*)d_in[3];
    const float* z0 = (const float*)d_in[4];
    int p = 5;
    const float* z_w        = (const float*)d_in[p++];
    const float* z_b        = (const float*)d_in[p++];
    const float* a_w        = (const float*)d_in[p++];
    const float* in_scale   = (const float*)d_in[p++];
    const float* in_bias    = (const float*)d_in[p++];
    const float* gru_wi     = (const float*)d_in[p++];
    const float* gru_bi     = (const float*)d_in[p++];
    const float* gru_wh     = (const float*)d_in[p++];
    const float* gru_bh     = (const float*)d_in[p++];
    const float* prior_h_w  = (const float*)d_in[p++];
    const float* prior_h_b  = (const float*)d_in[p++];
    const float* prior_scale= (const float*)d_in[p++];
    const float* prior_bias = (const float*)d_in[p++];
    const float* prior_w    = (const float*)d_in[p++];
    const float* prior_b    = (const float*)d_in[p++];
    const float* post_h_w   = (const float*)d_in[p++];
    const float* post_h_b   = (const float*)d_in[p++];
    const float* post_e_w   = (const float*)d_in[p++];
    const float* post_scale = (const float*)d_in[p++];
    const float* post_bias  = (const float*)d_in[p++];
    const float* post_w     = (const float*)d_in[p++];
    const float* post_b     = (const float*)d_in[p++];
    (void)in_sizes; (void)n_in; (void)out_size;

    float* out = (float*)d_out;
    float* priors_o   = out;
    float* posts_o    = out + (size_t)TB_DIM * SK_DIM;
    float* samples_o  = out + (size_t)2 * TB_DIM * SK_DIM;
    float* features_o = out + (size_t)3 * TB_DIM * SK_DIM;

    char* ws = (char*)d_ws;
    size_t cur = 0;
    auto take = [&](size_t bytes) { char* r = ws + cur; cur = align256(cur + bytes); return (void*)r; };
    bf16_t* zW_t   = (bf16_t*)take((size_t)HP_DIM * HP_DIM * 2);
    bf16_t* giW_t  = (bf16_t*)take((size_t)3 * D_DIM * HP_DIM * 2);
    bf16_t* ghW_t  = (bf16_t*)take((size_t)3 * D_DIM * D_DIM * 2);
    bf16_t* phW_t  = (bf16_t*)take((size_t)HP_DIM * D_DIM * 2);
    bf16_t* peW_t  = (bf16_t*)take((size_t)HP_DIM * E_DIM * 2);
    bf16_t* poW_t  = (bf16_t*)take((size_t)SK_DIM * HP_DIM * 2);
    bf16_t* prhW_t = (bf16_t*)take((size_t)HP_DIM * D_DIM * 2);
    bf16_t* prW_t  = (bf16_t*)take((size_t)SK_DIM * HP_DIM * 2);
    bf16_t* e_bf   = (bf16_t*)take((size_t)TB_DIM * E_DIM * 2);
    const size_t PSTRIDE = (size_t)B_DIM * 3 * D_DIM;
    float*  part   = (float*) take(8 * PSTRIDE * 4);
    bf16_t* za_bf  = (bf16_t*)take((size_t)B_DIM * HP_DIM * 2);
    float*  h      = (float*) take((size_t)B_DIM * D_DIM * 4);
    bf16_t* h_bf   = (bf16_t*)take((size_t)B_DIM * D_DIM * 2);
    bf16_t* z_bf   = (bf16_t*)take((size_t)B_DIM * SK_DIM * 2);
    bf16_t* pin_bf = (bf16_t*)take((size_t)B_DIM * HP_DIM * 2);
    bf16_t* hall_bf= (bf16_t*)take((size_t)TB_DIM * D_DIM * 2);
    float*  xp     = (float*) take((size_t)TB_DIM * HP_DIM * 4);
    bf16_t* xp_bf  = (bf16_t*)take((size_t)TB_DIM * HP_DIM * 2);
    (void)ws_size;

    const int TPB = 256;
    auto grid1d = [](size_t n) { return (int)((n + 255) / 256); };

    convert_weight_t<<<grid1d((size_t)HP_DIM*HP_DIM), TPB, 0, stream>>>(z_w,      zW_t,   SK_DIM, H_DIM, HP_DIM, HP_DIM);
    convert_weight_t<<<grid1d((size_t)3*D_DIM*HP_DIM), TPB, 0, stream>>>(gru_wi,  giW_t,  H_DIM, 3*D_DIM, HP_DIM, 3*D_DIM);
    convert_weight_t<<<grid1d((size_t)3*D_DIM*D_DIM), TPB, 0, stream>>>(gru_wh,   ghW_t,  D_DIM, 3*D_DIM, D_DIM, 3*D_DIM);
    convert_weight_t<<<grid1d((size_t)HP_DIM*D_DIM), TPB, 0, stream>>>(post_h_w,  phW_t,  D_DIM, H_DIM, D_DIM, HP_DIM);
    convert_weight_t<<<grid1d((size_t)HP_DIM*E_DIM), TPB, 0, stream>>>(post_e_w,  peW_t,  E_DIM, H_DIM, E_DIM, HP_DIM);
    convert_weight_t<<<grid1d((size_t)SK_DIM*HP_DIM), TPB, 0, stream>>>(post_w,   poW_t,  H_DIM, SK_DIM, HP_DIM, SK_DIM);
    convert_weight_t<<<grid1d((size_t)HP_DIM*D_DIM), TPB, 0, stream>>>(prior_h_w, prhW_t, D_DIM, H_DIM, D_DIM, HP_DIM);
    convert_weight_t<<<grid1d((size_t)SK_DIM*HP_DIM), TPB, 0, stream>>>(prior_w,  prW_t,  H_DIM, SK_DIM, HP_DIM, SK_DIM);
    convert_f32_bf16<<<grid1d((size_t)TB_DIM*E_DIM), TPB, 0, stream>>>(embeds, e_bf, TB_DIM*E_DIM);
    copy_f32<<<grid1d((size_t)B_DIM*D_DIM), TPB, 0, stream>>>(h0, h, B_DIM*D_DIM);

    for (int t = 0; t < T_DIM; ++t) {
        const float* z_src = (t == 0) ? z0 : (samples_o + (size_t)(t - 1) * B_DIM * SK_DIM);
        float* feat_t = features_o + (size_t)t * B_DIM * FEAT_DIM;
        float* post_t = posts_o   + (size_t)t * B_DIM * SK_DIM;
        float* samp_t = samples_o + (size_t)t * B_DIM * SK_DIM;

        prep_step<<<grid1d((size_t)B_DIM*D_DIM), TPB, 0, stream>>>(
            reset + (size_t)t * B_DIM, h, h_bf, z_src, z_bf);

        launch_gemm_split(z_bf, HP_DIM, zW_t, HP_DIM, part, PSTRIDE, 0, 4,
                          HP_DIM, HP_DIM, HP_DIM, B_DIM, stream);
        ln_elu_bf16<<<B_DIM, TPB, 0, stream>>>(part, 4, PSTRIDE, HP_DIM, z_b, H_DIM,
                                               in_scale, in_bias, za_bf, HP_DIM,
                                               action + (size_t)t * B_DIM * A_DIM, a_w);
        launch_gemm_split(za_bf, HP_DIM, giW_t, HP_DIM, part, PSTRIDE, 0, 4,
                          3*D_DIM, 3*D_DIM, HP_DIM, B_DIM, stream);
        launch_gemm_split(h_bf, D_DIM, ghW_t, D_DIM, part, PSTRIDE, 4, 4,
                          3*D_DIM, 3*D_DIM, D_DIM, B_DIM, stream);
        gru_gates<<<grid1d((size_t)B_DIM*D_DIM), TPB, 0, stream>>>(
            part, PSTRIDE, gru_bi, gru_bh, h, h_bf, feat_t);
        launch_gemm_split(h_bf, D_DIM, phW_t, D_DIM, part, PSTRIDE, 0, 4,
                          HP_DIM, HP_DIM, D_DIM, B_DIM, stream);
        launch_gemm_split(e_bf + (size_t)t * B_DIM * E_DIM, E_DIM, peW_t, E_DIM,
                          part, PSTRIDE, 4, 4, HP_DIM, HP_DIM, E_DIM, B_DIM, stream);
        ln_elu_bf16<<<B_DIM, TPB, 0, stream>>>(part, 8, PSTRIDE, HP_DIM, post_h_b, H_DIM,
                                               post_scale, post_bias, pin_bf, HP_DIM,
                                               nullptr, nullptr);
        launch_gemm_split(pin_bf, HP_DIM, poW_t, HP_DIM, part, PSTRIDE, 0, 4,
                          SK_DIM, SK_DIM, HP_DIM, B_DIM, stream);
        sum_bias<<<grid1d((size_t)B_DIM*SK_DIM), TPB, 0, stream>>>(part, PSTRIDE,
                                                                   post_b, post_t);
        gumbel_sample<<<(B_DIM * 32 + 7) / 8, TPB, 0, stream>>>(post_t, t, samp_t, feat_t);
    }

    gather_h_bf16<<<grid1d((size_t)TB_DIM*D_DIM), TPB, 0, stream>>>(features_o, hall_bf);
    launch_gemm(hall_bf, D_DIM, prhW_t, D_DIM, prior_h_b, xp, HP_DIM,
                TB_DIM, H_DIM, HP_DIM, D_DIM, stream);
    ln_elu_bf16<<<TB_DIM, TPB, 0, stream>>>(xp, 1, 0, HP_DIM, nullptr, H_DIM,
                                            prior_scale, prior_bias, xp_bf, HP_DIM,
                                            nullptr, nullptr);
    launch_gemm(xp_bf, HP_DIM, prW_t, HP_DIM, prior_b, priors_o, SK_DIM,
                TB_DIM, SK_DIM, SK_DIM, HP_DIM, stream);
}
